// RNNTDecoder_30039001269069
// MI455X (gfx1250) — compile-verified, matrix operand order checked by reading, not run on hardware
//
#include <hip/hip_runtime.h>
#include <hip/hip_bf16.h>
#include <math.h>

// Problem constants (match reference)
#define H   1024
#define V   5000
#define NL  2
#define B   2
#define T   128
#define U   64
#define H4  (4 * H)       // 4096
#define NOUT (V + 1)      // 5001
#define MJ  (B * T * U)   // 16384 joint rows

typedef unsigned short ushort_t;
typedef __attribute__((ext_vector_type(16))) __bf16 v16bf;
typedef __attribute__((ext_vector_type(8)))  float  v8f;

union FragU { v16bf v; uint4 q[2]; };

// ---------------------------------------------------------------------------
// fp32 -> bf16 (round to nearest even) flat convert
// ---------------------------------------------------------------------------
__global__ void cvt_f32_bf16(const float* __restrict__ in, ushort_t* __restrict__ out, int n) {
    int i = blockIdx.x * 256 + threadIdx.x;
    if (i < n) {
        unsigned u = __float_as_uint(in[i]);
        out[i] = (ushort_t)((u + 0x7FFFu + ((u >> 16) & 1u)) >> 16);
    }
}

__global__ void fill_f32(float* __restrict__ p, float v, int n) {
    int i = blockIdx.x * 256 + threadIdx.x;
    if (i < n) p[i] = v;
}

__global__ void bias_sum_kernel(const float* __restrict__ a, const float* __restrict__ b,
                                float* __restrict__ o, int n) {
    int i = blockIdx.x * 256 + threadIdx.x;
    if (i < n) o[i] = a[i] + b[i];
}

// x[b][u][:] = embed[targets[b][u]][:]
__global__ void embed_lookup(const float* __restrict__ embed, const int* __restrict__ targets,
                             float* __restrict__ x, int n) {
    int i = blockIdx.x * 256 + threadIdx.x;
    if (i >= n) return;
    int row = i >> 10;          // (b*U+u)
    int j   = i & (H - 1);
    x[i] = embed[(size_t)targets[row] * H + j];
}

// ---------------------------------------------------------------------------
// Tiled bf16 WMMA GEMM:  C[M,N](f32,row-major,ldc) = A[M,K](bf16) * B[N,K]^T(bf16) + bias[N]
// Block = 256 threads = 8 wave32, block tile 256x128.
// Wave tile 64x64 = 4x4 16x16 fragments -> 16 v_wmma per 16 b128 loads per K-step.
// K stepped by 32 -> v_wmma_f32_16x16x32_bf16 (f32 accumulate).
// ---------------------------------------------------------------------------
__global__ void __launch_bounds__(256)
wmma_gemm_bf16(const ushort_t* __restrict__ Abits, const ushort_t* __restrict__ Bbits,
               float* __restrict__ C, const float* __restrict__ bias,
               int M, int N, int K, int ldc)
{
    const int lane    = threadIdx.x & 31;
    const int wid     = threadIdx.x >> 5;
    const int wm      = wid & 3;   // 4 waves along M (64 rows each)
    const int wn      = wid >> 2;  // 2 waves along N (64 cols each)
    const int blockM  = blockIdx.y * 256;
    const int blockN  = blockIdx.x * 128;
    const int lane_lo = lane & 15;
    const int kofs    = (lane < 16) ? 0 : 8;

    const __bf16* A  = (const __bf16*)Abits;
    const __bf16* Bm = (const __bf16*)Bbits;

    v8f acc[4][4] = {};

    int arow[4], brow[4];
#pragma unroll
    for (int i = 0; i < 4; ++i) {
        int r = blockM + wm * 64 + i * 16 + lane_lo;
        arow[i] = (r < M) ? r : (M - 1);
    }
#pragma unroll
    for (int j = 0; j < 4; ++j) {
        int r = blockN + wn * 64 + j * 16 + lane_lo;
        brow[j] = (r < N) ? r : (N - 1);
    }

    for (int kk = 0; kk < K; kk += 32) {
        v16bf af[4], bfr[4];
#pragma unroll
        for (int i = 0; i < 4; ++i) {
            const __bf16* p = A + (size_t)arow[i] * K + kk + kofs;
            FragU f;
            f.q[0] = *(const uint4*)(p);        // K = kofs .. kofs+7
            f.q[1] = *(const uint4*)(p + 16);   // K = kofs+16 .. kofs+23
            af[i] = f.v;
        }
#pragma unroll
        for (int j = 0; j < 4; ++j) {
            const __bf16* p = Bm + (size_t)brow[j] * K + kk + kofs;
            FragU f;
            f.q[0] = *(const uint4*)(p);
            f.q[1] = *(const uint4*)(p + 16);
            bfr[j] = f.v;
        }
#pragma unroll
        for (int i = 0; i < 4; ++i)
#pragma unroll
            for (int j = 0; j < 4; ++j)
                acc[i][j] = __builtin_amdgcn_wmma_f32_16x16x32_bf16(
                    false, af[i], false, bfr[j], (short)0, acc[i][j], false, false);
    }

    // C/D layout: lanes 0-15 hold M=r (VGPR r), lanes 16-31 hold M=r+8; N = lane&15
#pragma unroll
    for (int i = 0; i < 4; ++i) {
        int mbase = blockM + wm * 64 + i * 16 + ((lane < 16) ? 0 : 8);
#pragma unroll
        for (int j = 0; j < 4; ++j) {
            int n = blockN + wn * 64 + j * 16 + lane_lo;
            if (n >= N) continue;
            float bv = bias ? bias[n] : 0.0f;
#pragma unroll
            for (int r = 0; r < 8; ++r) {
                int m = mbase + r;
                if (m < M) C[(size_t)m * ldc + n] = acc[i][j][r] + bv;
            }
        }
    }
}

// ---------------------------------------------------------------------------
// LSTM recurrence (fp32, latency-bound, Whh is L2-resident):
// one wave32 per output gate element: gates[b][g] = xg[b][t][g] + dot(Whh[g,:], h[b,:])
// ---------------------------------------------------------------------------
__global__ void __launch_bounds__(256)
lstm_gates_kernel(const float* __restrict__ xg, const float* __restrict__ Whh,
                  const float* __restrict__ h_prev, float* __restrict__ gates, int t)
{
    int gw   = (blockIdx.x * blockDim.x + threadIdx.x) >> 5; // global wave id, 0..B*4H-1
    int lane = threadIdx.x & 31;
    int g = gw & (H4 - 1);
    int b = gw >> 12;                  // H4 == 4096 == 2^12
    const float* wrow = Whh + (size_t)g * H;
    const float* hrow = h_prev + (size_t)b * H;
    float acc = 0.0f;
    for (int k = lane; k < H; k += 32) acc += wrow[k] * hrow[k];
#pragma unroll
    for (int off = 16; off > 0; off >>= 1) acc += __shfl_xor(acc, off, 32);
    if (lane == 0) gates[(size_t)b * H4 + g] = acc + xg[((size_t)b * U + t) * H4 + g];
}

__device__ inline float sigmoidf_(float x) { return 1.0f / (1.0f + __expf(-x)); }

// elementwise cell update; writes h into state and into layer output x_out[b][t][:]
__global__ void lstm_cell_kernel(const float* __restrict__ gates,
                                 float* __restrict__ h_state, float* __restrict__ c_state,
                                 float* __restrict__ x_out, int t)
{
    int idx = blockIdx.x * 256 + threadIdx.x;  // B*H
    if (idx >= B * H) return;
    int b = idx >> 10, j = idx & (H - 1);
    const float* gb = gates + (size_t)b * H4;
    float ig = sigmoidf_(gb[j]);
    float fg = sigmoidf_(gb[H + j]);
    float gg = tanhf(gb[2 * H + j]);
    float og = sigmoidf_(gb[3 * H + j]);
    float c = fg * c_state[idx] + ig * gg;
    float h = og * tanhf(c);
    c_state[idx] = c;
    h_state[idx] = h;
    x_out[((size_t)b * U + t) * H + j] = h;
}

// ---------------------------------------------------------------------------
// joint_bf16[m][h] = bf16( relu(enc_p[b][t][h] + prd_p[b][u][h]) ),  m = (b*T+t)*U+u
// ---------------------------------------------------------------------------
__global__ void joint_kernel(const float* __restrict__ enc_p, const float* __restrict__ prd_p,
                             ushort_t* __restrict__ joint)
{
    int m = blockIdx.x;
    int b = m / (T * U);
    int rem = m - b * (T * U);
    int t = rem / U;
    int u = rem - t * U;
    const float* ep = enc_p + ((size_t)b * T + t) * H;
    const float* pp = prd_p + ((size_t)b * U + u) * H;
    ushort_t* out = joint + (size_t)m * H;
    for (int j = threadIdx.x; j < H; j += 256) {
        float v = ep[j] + pp[j];
        v = v > 0.0f ? v : 0.0f;
        unsigned uu = __float_as_uint(v);
        out[j] = (ushort_t)((uu + 0x7FFFu + ((uu >> 16) & 1u)) >> 16);
    }
}

// ---------------------------------------------------------------------------
extern "C" void kernel_launch(void* const* d_in, const int* in_sizes, int n_in,
                              void* d_out, int out_size, void* d_ws, size_t ws_size,
                              hipStream_t stream)
{
    const float* encoder_out = (const float*)d_in[0];
    const int*   targets     = (const int*)d_in[1];
    const float* embed       = (const float*)d_in[2];
    const float* lstm_Wih    = (const float*)d_in[3];
    const float* lstm_Whh    = (const float*)d_in[4];
    const float* lstm_bih    = (const float*)d_in[5];
    const float* lstm_bhh    = (const float*)d_in[6];
    const float* enc_W       = (const float*)d_in[7];
    const float* enc_b       = (const float*)d_in[8];
    const float* pred_W      = (const float*)d_in[9];
    const float* pred_b      = (const float*)d_in[10];
    const float* out_W       = (const float*)d_in[11];
    const float* out_b       = (const float*)d_in[12];
    float* logits = (float*)d_out;
    (void)in_sizes; (void)n_in; (void)out_size; (void)ws_size;

    // workspace carve-up (~61 MB)
    char* base = (char*)d_ws;
    size_t off = 0;
    auto alloc = [&](size_t bytes) -> char* {
        char* p = base + off;
        off += (bytes + 255) & ~(size_t)255;
        return p;
    };
    ushort_t* outW_bf   = (ushort_t*)alloc((size_t)NOUT * H * 2);
    ushort_t* encW_bf   = (ushort_t*)alloc((size_t)H * H * 2);
    ushort_t* predW_bf  = (ushort_t*)alloc((size_t)H * H * 2);
    ushort_t* encout_bf = (ushort_t*)alloc((size_t)B * T * H * 2);
    ushort_t* Wih_bf    = (ushort_t*)alloc((size_t)H4 * H * 2);
    float*    x_a       = (float*)alloc((size_t)B * U * H * 4);
    float*    x_b       = (float*)alloc((size_t)B * U * H * 4);
    ushort_t* x_bf      = (ushort_t*)alloc((size_t)B * U * H * 2);
    float*    xg        = (float*)alloc((size_t)B * U * H4 * 4);
    float*    bias4h    = (float*)alloc((size_t)H4 * 4);
    float*    gates     = (float*)alloc((size_t)B * H4 * 4);
    float*    h_state   = (float*)alloc((size_t)B * H * 4);
    float*    c_state   = (float*)alloc((size_t)B * H * 4);
    float*    enc_p     = (float*)alloc((size_t)B * T * H * 4);
    float*    prd_p     = (float*)alloc((size_t)B * U * H * 4);
    ushort_t* joint_bf  = (ushort_t*)alloc((size_t)MJ * H * 2);

    auto cgrid = [](int n) { return dim3((unsigned)((n + 255) / 256)); };

    // weight / activation conversions to bf16
    cvt_f32_bf16<<<cgrid(NOUT * H), 256, 0, stream>>>(out_W, outW_bf, NOUT * H);
    cvt_f32_bf16<<<cgrid(H * H),    256, 0, stream>>>(enc_W, encW_bf, H * H);
    cvt_f32_bf16<<<cgrid(H * H),    256, 0, stream>>>(pred_W, predW_bf, H * H);
    cvt_f32_bf16<<<cgrid(B * T * H),256, 0, stream>>>(encoder_out, encout_bf, B * T * H);

    // prediction network: embedding + 2-layer LSTM
    embed_lookup<<<cgrid(B * U * H), 256, 0, stream>>>(embed, targets, x_a, B * U * H);

    float* xin = x_a;
    float* xout = x_b;
    for (int l = 0; l < NL; ++l) {
        cvt_f32_bf16<<<cgrid(H4 * H), 256, 0, stream>>>(lstm_Wih + (size_t)l * H4 * H, Wih_bf, H4 * H);
        bias_sum_kernel<<<cgrid(H4), 256, 0, stream>>>(lstm_bih + (size_t)l * H4,
                                                       lstm_bhh + (size_t)l * H4, bias4h, H4);
        cvt_f32_bf16<<<cgrid(B * U * H), 256, 0, stream>>>(xin, x_bf, B * U * H);
        // xg[B*U, 4H] = x * Wih^T + (bih+bhh)   (M=128, N=4096, K=1024)
        wmma_gemm_bf16<<<dim3(H4 / 128, 1), 256, 0, stream>>>(
            x_bf, Wih_bf, xg, bias4h, B * U, H4, H, H4);

        fill_f32<<<cgrid(B * H), 256, 0, stream>>>(h_state, 0.0f, B * H);
        fill_f32<<<cgrid(B * H), 256, 0, stream>>>(c_state, 0.0f, B * H);

        const float* Whh = lstm_Whh + (size_t)l * H4 * H;
        for (int t = 0; t < U; ++t) {
            // B*4H waves, 32 threads each -> 1024 blocks of 256
            lstm_gates_kernel<<<dim3((B * H4 * 32) / 256), 256, 0, stream>>>(xg, Whh, h_state, gates, t);
            lstm_cell_kernel<<<cgrid(B * H), 256, 0, stream>>>(gates, h_state, c_state, xout, t);
        }
        float* tmp = xin; xin = xout; xout = tmp;
    }
    // xin now holds pred_out [B,U,H]

    // joint network projections
    cvt_f32_bf16<<<cgrid(B * U * H), 256, 0, stream>>>(xin, x_bf, B * U * H);
    wmma_gemm_bf16<<<dim3(H / 128, 1), 256, 0, stream>>>(
        encout_bf, encW_bf, enc_p, enc_b, B * T, H, H, H);
    wmma_gemm_bf16<<<dim3(H / 128, 1), 256, 0, stream>>>(
        x_bf, predW_bf, prd_p, pred_b, B * U, H, H, H);

    // joint = relu(enc_p[:,:,None,:] + prd_p[:,None,:,:]) -> bf16 [MJ, H]
    joint_kernel<<<dim3(MJ), 256, 0, stream>>>(enc_p, prd_p, joint_bf);

    // logits [MJ, 5001] = joint * out_W^T + out_b   (the 168 GFLOP GEMM)
    wmma_gemm_bf16<<<dim3((NOUT + 127) / 128, MJ / 256), 256, 0, stream>>>(
        joint_bf, outW_bf, logits, out_b, MJ, NOUT, H, NOUT);
}
